// MultiHeadAttention_26946624815245
// MI455X (gfx1250) — matrix operand order repaired and checked by reading, not compiled, not run on hardware
//
#include <hip/hip_runtime.h>

typedef __attribute__((ext_vector_type(16))) _Float16 v16h;
typedef __attribute__((ext_vector_type(8)))  float    v8f;
typedef __attribute__((ext_vector_type(4)))  _Float16 h4;

union FragU {
    v16h  v;
    uint4 q[2];
};

// Low 32 bits of a generic pointer to __shared__ data are the LDS byte offset
// (CDNA5 ISA 10.2).
static __device__ __forceinline__ unsigned lds_off(const void* p) {
    return (unsigned)(unsigned long long)p;
}

// Async DMA: global memory -> LDS, 16 bytes per lane, tracked by ASYNCcnt.
static __device__ __forceinline__ void async_copy_b128(unsigned lds_byte_off,
                                                       const void* gaddr) {
    asm volatile("global_load_async_to_lds_b128 %0, %1, off"
                 :
                 : "v"(lds_byte_off), "v"((unsigned long long)gaddr)
                 : "memory");
}

static __device__ __forceinline__ void wait_async0() {
    asm volatile("s_wait_asynccnt 0x0" ::: "memory");
}

// ---------------------------------------------------------------------------
// f32 -> f16 cast (vectorized x4)
// ---------------------------------------------------------------------------
__global__ __launch_bounds__(256) void cvt_f32_to_f16(const float* __restrict__ in,
                                                      _Float16* __restrict__ out,
                                                      int n4) {
    int i = blockIdx.x * blockDim.x + threadIdx.x;
    if (i < n4) {
        float4 f = ((const float4*)in)[i];
        h4 o;
        o.x = (_Float16)f.x;
        o.y = (_Float16)f.y;
        o.z = (_Float16)f.z;
        o.w = (_Float16)f.w;
        ((h4*)out)[i] = o;
    }
}

// ---------------------------------------------------------------------------
// C[M,N] = A[M,K] @ W[N,K]^T + bias[N]
// 128x128 macrotile, BK=32, 8 waves (2x4), 64x32 per wave.
// Double-buffered LDS, global->LDS via async DMA (ASYNCcnt).
// ---------------------------------------------------------------------------
#define BM   128
#define BN   128
#define BK   32
#define LDSK 40   // 32 + 8 pad halfs -> 80B row stride (16B aligned)

template <typename OutT>
__global__ __launch_bounds__(256) void gemm_bias_wmma(
    const _Float16* __restrict__ A,    // [M,K] row-major
    const _Float16* __restrict__ W,    // [N,K] row-major (i.e. B^T)
    const float*    __restrict__ bias, // [N]
    OutT* __restrict__ Cout,           // [M,N]
    int M, int N, int K)
{
    __shared__ __align__(16) _Float16 As[2][BM * LDSK];
    __shared__ __align__(16) _Float16 Bs[2][BN * LDSK];

    const int tid  = threadIdx.x;
    const int lane = tid & 31;
    const int wv   = tid >> 5;
    const int l16  = lane & 15;
    const int half = lane >> 4;
    const int wm   = (wv >> 2) * 64;   // wave M offset in tile (0/64)
    const int wn   = (wv & 3) * 32;    // wave N offset in tile (0..96)
    const int m0   = blockIdx.y * BM;
    const int n0   = blockIdx.x * BN;

    // This thread's two staging chunks (row, 8-half column group)
    const int row0 = tid >> 2;                // chunks 0..255
    const int cq0  = tid & 3;
    const int row1 = (tid + 256) >> 2;        // chunks 256..511
    const int cq1  = (tid + 256) & 3;

    v8f acc[4][2];
#pragma unroll
    for (int i = 0; i < 4; ++i)
#pragma unroll
        for (int j = 0; j < 2; ++j)
            acc[i][j] = (v8f){0.f, 0.f, 0.f, 0.f, 0.f, 0.f, 0.f, 0.f};

    auto stage = [&](int k0, int buf) {
        async_copy_b128(lds_off(&As[buf][row0 * LDSK + cq0 * 8]),
                        &A[(size_t)(m0 + row0) * K + k0 + cq0 * 8]);
        async_copy_b128(lds_off(&Bs[buf][row0 * LDSK + cq0 * 8]),
                        &W[(size_t)(n0 + row0) * K + k0 + cq0 * 8]);
        async_copy_b128(lds_off(&As[buf][row1 * LDSK + cq1 * 8]),
                        &A[(size_t)(m0 + row1) * K + k0 + cq1 * 8]);
        async_copy_b128(lds_off(&Bs[buf][row1 * LDSK + cq1 * 8]),
                        &W[(size_t)(n0 + row1) * K + k0 + cq1 * 8]);
    };

    stage(0, 0);
    const int nk = K / BK;
    for (int ki = 0; ki < nk; ++ki) {
        const int buf = ki & 1;
        wait_async0();     // our DMA for `buf` has landed in LDS
        __syncthreads();   // everyone's has; everyone done reading buf^1
        if (ki + 1 < nk) stage((ki + 1) * BK, buf ^ 1);  // overlaps compute

        FragU aF[4], bF[2];
#pragma unroll
        for (int i = 0; i < 4; ++i) {
            int r = wm + i * 16 + l16;
            aF[i].q[0] = *(const uint4*)&As[buf][r * LDSK + half * 8];
            aF[i].q[1] = *(const uint4*)&As[buf][r * LDSK + 16 + half * 8];
        }
#pragma unroll
        for (int j = 0; j < 2; ++j) {
            int c = wn + j * 16 + l16;
            bF[j].q[0] = *(const uint4*)&Bs[buf][c * LDSK + half * 8];
            bF[j].q[1] = *(const uint4*)&Bs[buf][c * LDSK + 16 + half * 8];
        }
#pragma unroll
        for (int i = 0; i < 4; ++i)
#pragma unroll
            for (int j = 0; j < 2; ++j)
                acc[i][j] = __builtin_amdgcn_wmma_f32_16x16x32_f16(
                    false, aF[i].v, false, bF[j].v, (short)0, acc[i][j],
                    false, false);
    }

    // Epilogue: C/D layout -> lane holds col=l16, rows vj + half*8.
#pragma unroll
    for (int j = 0; j < 2; ++j) {
        int gn = n0 + wn + j * 16 + l16;
        float bv = bias[gn];
#pragma unroll
        for (int i = 0; i < 4; ++i) {
#pragma unroll
            for (int vj = 0; vj < 8; ++vj) {
                int gm = m0 + wm + i * 16 + vj + half * 8;
                float val = acc[i][j][vj] + bv;
                Cout[(size_t)gm * N + gn] = (OutT)val;
            }
        }
    }
}

// ---------------------------------------------------------------------------
// Causal flash attention over f16 qkv [B,T,3C]; writes merged-head y [B,T,C].
// Block = 4 waves = 64 q rows; wave owns 16 q rows. KV chunks of 32.
// K chunk staged once per block via async DMA (was 4x redundant per wave);
// V chunk transposed into LDS; softmax denominator via L += P @ ones WMMA.
// ---------------------------------------------------------------------------
__global__ __launch_bounds__(128) void attn_wmma(const _Float16* __restrict__ qkv,
                                                 _Float16* __restrict__ Y)
{
    constexpr int T  = 2048;
    constexpr int C  = 1024;
    constexpr int DK = 64;
    constexpr int C3 = 3 * C;
    constexpr int KRS = 72;   // Kr row stride in halfs (144B, 16B aligned)

    __shared__ __align__(16) _Float16 Kr[32 * KRS];   // K chunk, row-major
    __shared__ __align__(16) _Float16 Vt[DK][40];     // V^T: [d][kv]
    __shared__ __align__(16) _Float16 Pb[4][16][40];  // per-wave P tile

    const int tid  = threadIdx.x;
    const int lane = tid & 31;
    const int w    = tid >> 5;
    const int l16  = lane & 15;
    const int half = lane >> 4;

    const int q0 = blockIdx.x * 64;
    const int b  = blockIdx.y >> 4;
    const int h  = blockIdx.y & 15;
    const size_t bT = (size_t)b * T;

    const int qw = q0 + w * 16;  // this wave's first q row

    // Q fragments: 16 rows x 64 d -> two 16x32 A-fragments, loaded once.
    FragU qA[2];
    {
        const _Float16* qp = qkv + (bT + qw + l16) * C3 + h * DK;
#pragma unroll
        for (int c = 0; c < 2; ++c) {
            qA[c].q[0] = *(const uint4*)(qp + c * 32 + half * 8);
            qA[c].q[1] = *(const uint4*)(qp + c * 32 + 16 + half * 8);
        }
    }

    // Constant ones B-fragment for the L += P @ 1 trick.
    FragU onesF;
#pragma unroll
    for (int e = 0; e < 16; ++e) onesF.v[e] = (_Float16)1.0f;

    v8f accY[4];
#pragma unroll
    for (int dt = 0; dt < 4; ++dt)
        accY[dt] = (v8f){0.f, 0.f, 0.f, 0.f, 0.f, 0.f, 0.f, 0.f};
    v8f accL = (v8f){0.f, 0.f, 0.f, 0.f, 0.f, 0.f, 0.f, 0.f};

    float Mrow[8];
#pragma unroll
    for (int vj = 0; vj < 8; ++vj) Mrow[vj] = -1e30f;

    // This thread's two K-staging chunks: 32 rows x 128B = 256 x 16B chunks.
    const int krow0 = tid >> 3, kcq0 = tid & 7;
    const int krow1 = (tid + 128) >> 3, kcq1 = (tid + 128) & 7;

    const int nchunks = (q0 >> 5) + 2;
    for (int kc = 0; kc < nchunks; ++kc) {
        const int kv0 = kc * 32;

        __syncthreads();  // all waves done reading previous Kr/Vt

        // Async-stage K chunk (32 kv rows x 64 d) into LDS, once per block.
        async_copy_b128(lds_off(&Kr[krow0 * KRS + kcq0 * 8]),
                        qkv + (bT + kv0 + krow0) * C3 + C + h * DK + kcq0 * 8);
        async_copy_b128(lds_off(&Kr[krow1 * KRS + kcq1 * 8]),
                        qkv + (bT + kv0 + krow1) * C3 + C + h * DK + kcq1 * 8);

        {   // stage V chunk transposed (overlaps the K DMA): 16 elems/thread
            int r  = tid >> 2;
            int dq = (tid & 3) * 16;
            const _Float16* vp =
                qkv + (bT + kv0 + r) * C3 + 2 * C + h * DK + dq;
            FragU tmp;
            tmp.q[0] = *(const uint4*)(vp);
            tmp.q[1] = *(const uint4*)(vp + 8);
#pragma unroll
            for (int e = 0; e < 16; ++e) Vt[dq + e][r] = tmp.v[e];
        }
        wait_async0();
        __syncthreads();

        // S = (Q @ K^T) * 1/sqrt(Dk), two 16x16 kv tiles, causal-masked
        float sc[2][8];
#pragma unroll
        for (int t = 0; t < 2; ++t) {
            v8f s = (v8f){0.f, 0.f, 0.f, 0.f, 0.f, 0.f, 0.f, 0.f};
            FragU kB[2];
            const _Float16* kp = &Kr[(t * 16 + l16) * KRS];
#pragma unroll
            for (int c = 0; c < 2; ++c) {
                kB[c].q[0] = *(const uint4*)(kp + c * 32 + half * 8);
                kB[c].q[1] = *(const uint4*)(kp + c * 32 + 16 + half * 8);
            }
            s = __builtin_amdgcn_wmma_f32_16x16x32_f16(
                false, qA[0].v, false, kB[0].v, (short)0, s, false, false);
            s = __builtin_amdgcn_wmma_f32_16x16x32_f16(
                false, qA[1].v, false, kB[1].v, (short)0, s, false, false);

            const int colg = kv0 + t * 16 + l16;
#pragma unroll
            for (int vj = 0; vj < 8; ++vj) {
                float sv = s[vj] * 0.125f;  // 1/sqrt(64)
                int rowg = qw + vj + half * 8;
                if (colg > rowg) sv = -1e30f;  // causal (finite sentinel)
                sc[t][vj] = sv;
            }
        }

        // Online softmax update; write P (f16) to this wave's LDS tile.
#pragma unroll
        for (int vj = 0; vj < 8; ++vj) {
            float m2 = fmaxf(sc[0][vj], sc[1][vj]);
            m2 = fmaxf(m2, __shfl_xor(m2, 1, 16));
            m2 = fmaxf(m2, __shfl_xor(m2, 2, 16));
            m2 = fmaxf(m2, __shfl_xor(m2, 4, 16));
            m2 = fmaxf(m2, __shfl_xor(m2, 8, 16));
            float Mn    = fmaxf(Mrow[vj], m2);
            float alpha = __expf(Mrow[vj] - Mn);
            Mrow[vj] = Mn;
#pragma unroll
            for (int dt = 0; dt < 4; ++dt) accY[dt][vj] *= alpha;
            accL[vj] *= alpha;
            float p0 = __expf(sc[0][vj] - Mn);
            float p1 = __expf(sc[1][vj] - Mn);
            Pb[w][vj + half * 8][l16]      = (_Float16)p0;
            Pb[w][vj + half * 8][16 + l16] = (_Float16)p1;
        }

        // P fragment from LDS (same-wave DS ops are in-order).
        FragU pA;
        pA.q[0] = *(const uint4*)&Pb[w][l16][half * 8];
        pA.q[1] = *(const uint4*)&Pb[w][l16][16 + half * 8];

        // L += P @ ones  (row sums of P, C-layout matches running state)
        accL = __builtin_amdgcn_wmma_f32_16x16x32_f16(
            false, pA.v, false, onesF.v, (short)0, accL, false, false);

        // y += P @ V
#pragma unroll
        for (int dt = 0; dt < 4; ++dt) {
            FragU vB;
            vB.q[0] = *(const uint4*)&Vt[dt * 16 + l16][half * 8];
            vB.q[1] = *(const uint4*)&Vt[dt * 16 + l16][16 + half * 8];
            accY[dt] = __builtin_amdgcn_wmma_f32_16x16x32_f16(
                false, pA.v, false, vB.v, (short)0, accY[dt], false, false);
        }
    }

    // Normalize and store y (merged heads) as f16
#pragma unroll
    for (int vj = 0; vj < 8; ++vj) {
        float inv = 1.0f / accL[vj];
        int trow = qw + vj + half * 8;
#pragma unroll
        for (int dt = 0; dt < 4; ++dt) {
            float yv = accY[dt][vj] * inv;
            Y[(bT + trow) * C + h * DK + dt * 16 + l16] = (_Float16)yv;
        }
    }
}

// ---------------------------------------------------------------------------
// Host-side orchestration
// ---------------------------------------------------------------------------
extern "C" void kernel_launch(void* const* d_in, const int* in_sizes, int n_in,
                              void* d_out, int out_size, void* d_ws, size_t ws_size,
                              hipStream_t stream) {
    const float* x     = (const float*)d_in[0];
    const float* W_in  = (const float*)d_in[1];
    const float* b_in  = (const float*)d_in[2];
    const float* W_out = (const float*)d_in[3];
    const float* b_out = (const float*)d_in[4];
    float* out = (float*)d_out;

    const int Tt = 2048, Cc = 1024, Hh = 16;
    const int Bb = in_sizes[0] / (Tt * Cc);   // = 4
    const int Mrows = Bb * Tt;                // = 8192

    // Workspace layout (f16): x | W_in | W_out | qkv | y  (~92 MB total)
    _Float16* xb   = (_Float16*)d_ws;
    _Float16* wib  = xb   + (size_t)Mrows * Cc;
    _Float16* wob  = wib  + (size_t)3 * Cc * Cc;
    _Float16* qkvb = wob  + (size_t)Cc * Cc;
    _Float16* yb   = qkvb + (size_t)Mrows * 3 * Cc;

    {   // casts
        size_t nx  = (size_t)Mrows * Cc;
        size_t nwi = (size_t)3 * Cc * Cc;
        size_t nwo = (size_t)Cc * Cc;
        cvt_f32_to_f16<<<(int)((nx / 4 + 255) / 256), 256, 0, stream>>>(x, xb, (int)(nx / 4));
        cvt_f32_to_f16<<<(int)((nwi / 4 + 255) / 256), 256, 0, stream>>>(W_in, wib, (int)(nwi / 4));
        cvt_f32_to_f16<<<(int)((nwo / 4 + 255) / 256), 256, 0, stream>>>(W_out, wob, (int)(nwo / 4));
    }

    // qkv = x @ W_in^T + b_in   -> f16 intermediate (L2-resident, 48 MB)
    gemm_bias_wmma<_Float16><<<dim3(3 * Cc / BN, Mrows / BM), 256, 0, stream>>>(
        xb, wib, b_in, qkvb, Mrows, 3 * Cc, Cc);

    // causal MHA -> y (f16, merged heads)
    attn_wmma<<<dim3(Tt / 64, Bb * Hh), 128, 0, stream>>>(qkvb, yb);

    // out = y @ W_out^T + b_out -> f32 output
    gemm_bias_wmma<float><<<dim3(Cc / BN, Mrows / BM), 256, 0, stream>>>(
        yb, wob, b_out, out, Mrows, Cc, Cc);
}
